// NeighborEmbedding_36146444763345
// MI455X (gfx1250) — matrix operand order repaired and checked by reading, not compiled, not run on hardware
//
#include <hip/hip_runtime.h>
#include <hip/hip_bf16.h>

typedef __attribute__((ext_vector_type(16))) _Float16 v16h;
typedef __attribute__((ext_vector_type(8)))  float    v8f;

#define NN      50000
#define EE      800000
#define CC      256
#define RBF     64
#define PI_OVER_CUTOFF 0.6283185307179586f   // pi / 5.0
#define CUTOFF_F 5.0f

// ---------------------------------------------------------------------------
// Pack a row-major f32 weight matrix [ktiles*32, 256] into f16 B-fragments.
// Fragment layout (per ISA 7.12.2, 16-bit B 32x16):
//   lane l holds N = l%16, K = kt*32 + (e/8)*16 + (l/16)*8 + e%8   for e = 0..15
//   frag[((kt*16 + ct)*32 + lane)*16 + e]
// ---------------------------------------------------------------------------
__global__ void pack_w_kernel(const float* __restrict__ W, _Float16* __restrict__ frag,
                              int total) {
    int idx = blockIdx.x * blockDim.x + threadIdx.x;
    if (idx >= total) return;
    int e    = idx & 15;
    int lane = (idx >> 4) & 31;
    int tile = idx >> 9;          // tile = kt*16 + ct
    int kt   = tile >> 4;
    int ct   = tile & 15;
    int K    = kt * 32 + (e >> 3) * 16 + (lane >> 4) * 8 + (e & 7);
    int col  = ct * 16 + (lane & 15);
    frag[idx] = (_Float16)W[K * CC + col];
}

// Load one A fragment (16x32 f16, ISA 7.12.2 layout) from a row-major f16 LDS
// tile with row stride `ldr` halves.  Per lane: two contiguous 8-half runs.
__device__ inline v16h load_a_frag(const _Float16* row_ptr) {
    // row_ptr already points at (row, k0) with k0 = ktBase + halfsel*8
    v16h a;
#pragma unroll
    for (int i = 0; i < 8; ++i) {
        a[i]     = row_ptr[i];
        a[i + 8] = row_ptr[i + 16];
    }
    return a;
}

// ---------------------------------------------------------------------------
// Edge phase: fused cutoff + (edge_feats @ W_dist + b_dist)*cut * x_j
//             + segment_sum into agg.
// Block = 512 threads = 16 waves, one block per 64-edge super-tile
// (4 M-tiles of 16 edges).  Wave w owns output columns [16w, 16w+16), K = 64.
// A tile is converted to f16 in LDS once per block; B fragments are loaded
// once per wave and reused across the 4 M-tiles.
// ---------------------------------------------------------------------------
__global__ __launch_bounds__(512) void edge_kernel(
    const int*   __restrict__ node_z,
    const int*   __restrict__ senders,
    const int*   __restrict__ receivers,
    const float* __restrict__ edge_weight,
    const float* __restrict__ edge_feats,
    const float* __restrict__ embed_table,
    const float* __restrict__ b_dist,
    const _Float16* __restrict__ wd_frag,
    float*       __restrict__ agg)
{
    __shared__ float    s_cut[64];
    __shared__ int      s_z[64];
    __shared__ int      s_recv[64];
    __shared__ _Float16 s_a[64 * RBF];     // 8 KB: 64 edge rows x 64 rbf, f16

    const int eBase = blockIdx.x * 64;
    const int t = threadIdx.x;

    if (t < 64) {
        int e = eBase + t;
        float w = edge_weight[e];
        float c = 0.5f * (__cosf(w * PI_OVER_CUTOFF) + 1.0f);
        s_cut[t]  = (w < CUTOFF_F) ? c : 0.0f;
        s_z[t]    = node_z[senders[e]];
        s_recv[t] = receivers[e];
    }
    // Convert A tile f32 -> f16 into LDS: 64 rows x 64 cols, 8 values/thread.
    {
        int row = t >> 3;
        int cb  = (t & 7) * 8;
        const float* src = edge_feats + (size_t)(eBase + row) * RBF + cb;
        _Float16* dst = s_a + row * RBF + cb;
#pragma unroll
        for (int i = 0; i < 8; ++i) dst[i] = (_Float16)src[i];
    }
    __syncthreads();

    const int wave    = t >> 5;        // 0..15 -> column tile
    const int lane    = t & 31;
    const int halfsel = lane >> 4;     // 0 or 1
    const int mrow    = lane & 15;     // A-row / C-column index
    const int n       = wave * 16 + mrow;
    const float bd    = b_dist[n];

    // B fragments for both K tiles, loaded once, reused across 4 M-tiles.
    v16h b0 = *(const v16h*)(wd_frag + ((size_t)(0 * 16 + wave) * 32 + lane) * 16);
    v16h b1 = *(const v16h*)(wd_frag + ((size_t)(1 * 16 + wave) * 32 + lane) * 16);

#pragma unroll
    for (int mt = 0; mt < 4; ++mt) {
        const int rb = mt * 16;                            // row base within block
        const _Float16* ap = s_a + (rb + mrow) * RBF + halfsel * 8;
        v16h a0 = load_a_frag(ap);        // K =  0..31 tile
        v16h a1 = load_a_frag(ap + 32);   // K = 32..63 tile
        v8f acc = {};
        acc = __builtin_amdgcn_wmma_f32_16x16x32_f16(false, a0, false, b0,
                                                     (short)0, acc, false, false);
        acc = __builtin_amdgcn_wmma_f32_16x16x32_f16(false, a1, false, b1,
                                                     (short)0, acc, false, false);

        // Epilogue: C layout -> lane holds M = rb + v + 8*halfsel, N = n.
        const bool uniform = (s_recv[rb] == s_recv[rb + 15]);  // receivers sorted
        if (uniform) {
            float r = 0.0f;
#pragma unroll
            for (int v = 0; v < 8; ++v) {
                int m = rb + halfsel * 8 + v;
                r += (acc[v] + bd) * s_cut[m] * embed_table[s_z[m] * CC + n];
            }
            r += __shfl_xor(r, 16, 32);                    // fold the two M halves
            if (halfsel == 0)
                atomicAdd(&agg[(size_t)s_recv[rb] * CC + n], r);
        } else {
#pragma unroll
            for (int v = 0; v < 8; ++v) {
                int m = rb + halfsel * 8 + v;
                float val = (acc[v] + bd) * s_cut[m] * embed_table[s_z[m] * CC + n];
                atomicAdd(&agg[(size_t)s_recv[m] * CC + n], val);
            }
        }
    }
}

// ---------------------------------------------------------------------------
// Node phase: out = concat(node_feats, agg) @ W_comb + b_comb.
// Block = 512 threads = 16 waves, one block per 80-node super-tile
// (5 M-tiles; 50000 = 625 * 80).  K = 512 -> 16 K-tiles; each B fragment is
// loaded once per wave per kt and reused across the 5 M-tiles (80 WMMA/wave).
// A (80 x 512 f16 = 80 KB) staged in dynamic LDS, converted once per block.
// ---------------------------------------------------------------------------
__global__ __launch_bounds__(512) void node_kernel(
    const float* __restrict__ node_feats,
    const float* __restrict__ agg,
    const _Float16* __restrict__ wc_frag,
    const float* __restrict__ b_comb,
    float*       __restrict__ out)
{
    extern __shared__ _Float16 s_a[];      // 80 rows x 512 halves = 80 KB

    const int t = threadIdx.x;
    const int nodeBase = blockIdx.x * 80;

    // Convert concat(node_feats, agg) rows to f16 in LDS.
    // Wave-per-row sweep: threads 0..31 cover one full 512-wide row (16 f32 each).
#pragma unroll
    for (int mt = 0; mt < 5; ++mt) {
        int row = mt * 16 + (t >> 5);
        int cb  = (t & 31) * 16;
        const float* src = (cb < CC)
            ? node_feats + (size_t)(nodeBase + row) * CC + cb
            : agg        + (size_t)(nodeBase + row) * CC + (cb - CC);
        _Float16* dst = s_a + row * 512 + cb;
#pragma unroll
        for (int i = 0; i < 16; ++i) dst[i] = (_Float16)src[i];
    }
    __syncthreads();

    const int wave    = t >> 5;
    const int lane    = t & 31;
    const int halfsel = lane >> 4;
    const int mrow    = lane & 15;
    const int n       = wave * 16 + mrow;

    v8f acc0 = {}, acc1 = {}, acc2 = {}, acc3 = {}, acc4 = {};
#pragma unroll
    for (int kt = 0; kt < 16; ++kt) {
        v16h b = *(const v16h*)(wc_frag + ((size_t)(kt * 16 + wave) * 32 + lane) * 16);
        const _Float16* ap = s_a + mrow * 512 + kt * 32 + halfsel * 8;
        acc0 = __builtin_amdgcn_wmma_f32_16x16x32_f16(false, load_a_frag(ap),
                                                      false, b, (short)0, acc0, false, false);
        acc1 = __builtin_amdgcn_wmma_f32_16x16x32_f16(false, load_a_frag(ap + 16 * 512),
                                                      false, b, (short)0, acc1, false, false);
        acc2 = __builtin_amdgcn_wmma_f32_16x16x32_f16(false, load_a_frag(ap + 32 * 512),
                                                      false, b, (short)0, acc2, false, false);
        acc3 = __builtin_amdgcn_wmma_f32_16x16x32_f16(false, load_a_frag(ap + 48 * 512),
                                                      false, b, (short)0, acc3, false, false);
        acc4 = __builtin_amdgcn_wmma_f32_16x16x32_f16(false, load_a_frag(ap + 64 * 512),
                                                      false, b, (short)0, acc4, false, false);
    }

    const float bc = b_comb[n];
    v8f accs[5] = {acc0, acc1, acc2, acc3, acc4};
#pragma unroll
    for (int mt = 0; mt < 5; ++mt) {
#pragma unroll
        for (int v = 0; v < 8; ++v) {
            int row = nodeBase + mt * 16 + halfsel * 8 + v;
            out[(size_t)row * CC + n] = accs[mt][v] + bc;
        }
    }
}

// ---------------------------------------------------------------------------
extern "C" void kernel_launch(void* const* d_in, const int* in_sizes, int n_in,
                              void* d_out, int out_size, void* d_ws, size_t ws_size,
                              hipStream_t stream) {
    const int*   node_z      = (const int*)  d_in[0];
    const float* node_feats  = (const float*)d_in[1];
    const int*   senders     = (const int*)  d_in[2];
    const int*   receivers   = (const int*)  d_in[3];
    const float* edge_weight = (const float*)d_in[4];
    const float* edge_feats  = (const float*)d_in[5];
    const float* embed_table = (const float*)d_in[6];
    const float* W_dist      = (const float*)d_in[7];
    const float* b_dist      = (const float*)d_in[8];
    const float* W_comb      = (const float*)d_in[9];
    const float* b_comb      = (const float*)d_in[10];
    float* out = (float*)d_out;

    // Workspace layout: agg [N*C f32], then f16 weight fragments.
    char* ws = (char*)d_ws;
    float* agg = (float*)ws;
    size_t aggBytes = (size_t)NN * CC * sizeof(float);          // 51.2 MB, 256B-aligned
    _Float16* wd_frag = (_Float16*)(ws + aggBytes);             // 64*256 halves = 32 KB
    _Float16* wc_frag = wd_frag + RBF * CC;                     // 512*256 halves = 256 KB

    hipMemsetAsync(agg, 0, aggBytes, stream);

    const int wd_total = 2  * 16 * 32 * 16;   // 16384
    const int wc_total = 16 * 16 * 32 * 16;   // 131072
    pack_w_kernel<<<(wd_total + 255) / 256, 256, 0, stream>>>(W_dist, wd_frag, wd_total);
    pack_w_kernel<<<(wc_total + 255) / 256, 256, 0, stream>>>(W_comb, wc_frag, wc_total);

    edge_kernel<<<EE / 64, 512, 0, stream>>>(node_z, senders, receivers, edge_weight,
                                             edge_feats, embed_table, b_dist,
                                             wd_frag, agg);

    const size_t node_lds = (size_t)80 * 512 * sizeof(_Float16);   // 80 KB
    node_kernel<<<NN / 80, 512, node_lds, stream>>>(node_feats, agg, wc_frag, b_comb, out);
}
